// MultiHeadMchAttnBlock_3092376453889
// MI455X (gfx1250) — matrix operand order repaired and checked by reading, not compiled, not run on hardware
//
#include <hip/hip_runtime.h>
#include <hip/hip_bf16.h>
#include <math.h>

// ---------------------------------------------------------------------------
// MultiHeadMchAttnBlock fused kernel for MI455X (gfx1250, wave32, WMMA).
// B=4, M=1024, ND=64, ED=8, D=64, H=4.  Memory-bound: comp_val = 128MB is the
// dominant stream (~6.4us floor @23.3TB/s).  comp_val tiles are double-
// buffered in LDS via global_load_async_to_lds_b128 (overlapped with compute)
// and shared by the 4 heads of a (b,k16) group.  attn @ Wh uses
// v_wmma_f32_16x16x32_bf16 (f32 accumulate); Wh is pre-stored transposed in
// bf16 so B fragments are raw b128 loads.
// ---------------------------------------------------------------------------

#ifndef __has_builtin
#define __has_builtin(x) 0
#endif

#if __has_builtin(__builtin_amdgcn_global_load_async_to_lds_b128)
#define ASYNC_LDS 1
#else
#define ASYNC_LDS 0
#endif

static constexpr int Bc = 4;     // batch
static constexpr int Mc = 1024;  // nodes
static constexpr int NDc = 64;   // node feat dim
static constexpr int EDc = 8;    // edge feat dim
static constexpr int Dc = 64;    // head dim
static constexpr int Hc = 4;     // heads
static constexpr int PSTR = 33;  // padded P-tile row stride (bank-conflict free)
static constexpr int NCHUNK = Mc / 32;  // 32 q-chunks
static constexpr float ALPHA = 0.2f;
static constexpr float NEGV = -9000000000000000.0f;

typedef __attribute__((ext_vector_type(16))) __bf16 v16bf;
typedef __attribute__((ext_vector_type(8)))  float  v8f;
typedef __attribute__((ext_vector_type(4)))  int    v4i;

typedef __attribute__((address_space(1))) v4i gv4i;  // global int4
typedef __attribute__((address_space(3))) v4i lv4i;  // LDS int4

__device__ __forceinline__ void copy16_g2lds(const void* gsrc, void* ldsdst) {
#if ASYNC_LDS
  __builtin_amdgcn_global_load_async_to_lds_b128(
      (gv4i*)gsrc, (lv4i*)ldsdst, 0, 0);
#else
  *(float4*)ldsdst = *(const float4*)gsrc;
#endif
}

template <int N>
__device__ __forceinline__ void wait_async_le() {
#if ASYNC_LDS
#if __has_builtin(__builtin_amdgcn_s_wait_asynccnt)
  __builtin_amdgcn_s_wait_asynccnt(N);
#else
  asm volatile("s_wait_asynccnt %0" ::"i"(N) : "memory");
#endif
#endif
}

// ---------------------------------------------------------------------------
// Kernel 1: Wh = h @ W stored transposed AND bf16: WhTbf[b,h,d,m].
// Wh1/Wh2 row biases in f32; w_e[h,e] = W_edge[h,e,:] . a3[h,:].
// ---------------------------------------------------------------------------
__global__ __launch_bounds__(256) void precompute_kernel(
    const float* __restrict__ hfeat, const float* __restrict__ W,
    const float* __restrict__ W_edge, const float* __restrict__ avec,
    __bf16* __restrict__ WhTbf, float* __restrict__ Wh1,
    float* __restrict__ Wh2, float* __restrict__ w_e) {
  int idx = blockIdx.x * blockDim.x + threadIdx.x;

  if (idx < Hc * EDc) {  // tiny w_e side computation
    int hh = idx >> 3, e = idx & 7;
    float s = 0.f;
    #pragma unroll 16
    for (int d = 0; d < Dc; ++d)
      s += W_edge[((size_t)hh * EDc + e) * Dc + d] * avec[hh * 3 * Dc + 2 * Dc + d];
    w_e[idx] = s;
  }

  if (idx >= Bc * Hc * Mc) return;
  int m  = idx & (Mc - 1);
  int hh = (idx / Mc) & (Hc - 1);
  int b  = idx / (Mc * Hc);

  float hrow[NDc];
  const float* hp = hfeat + ((size_t)b * Mc + m) * NDc;
  #pragma unroll 16
  for (int nd = 0; nd < NDc; ++nd) hrow[nd] = hp[nd];

  const float* Wp = W + (size_t)hh * NDc * Dc;
  const float* ap = avec + (size_t)hh * 3 * Dc;
  float wh1 = 0.f, wh2 = 0.f;
  __bf16* whtp = WhTbf + (size_t)(b * Hc + hh) * Dc * Mc;
  for (int d = 0; d < Dc; ++d) {
    float s = 0.f;
    #pragma unroll 16
    for (int nd = 0; nd < NDc; ++nd) s += hrow[nd] * Wp[nd * Dc + d];
    whtp[(size_t)d * Mc + m] = (__bf16)s;   // transposed + bf16: WMMA-B ready
    wh1 += s * ap[d];
    wh2 += s * ap[Dc + d];
  }
  Wh1[(size_t)(b * Hc + hh) * Mc + m] = wh1;
  Wh2[(size_t)(b * Hc + hh) * Mc + m] = wh2;
}

// ---------------------------------------------------------------------------
// Kernel 2: fused edge-score + leaky-relu + mask + online softmax + attn@Wh.
// Block = 256 threads = 8 waves = 2 groups of 4 waves.
// Group g handles (b, k0..k0+15); wave within group handles head h.
// Double-buffered async tile pipeline: chunk i+1 streams into LDS while
// chunk i is consumed.
// ---------------------------------------------------------------------------
__global__ __launch_bounds__(256, 1) void attn_main_kernel(
    const float* __restrict__ comp_val, const int* __restrict__ mch_mask,
    const __bf16* __restrict__ WhTbf, const float* __restrict__ Wh1,
    const float* __restrict__ Wh2, const float* __restrict__ w_e,
    float* __restrict__ out) {
  __shared__ float s_comp[2][2][16 * 32 * EDc];  // [group][buf] 16KB tiles
  __shared__ int   s_mask[2][2][16 * 32];        // [group][buf] 2KB tiles
  __shared__ float s_p[8][16 * PSTR];            // per-wave e/P staging, padded
  __shared__ float s_scale[8][16];               // per-wave softmax rescale
  __shared__ float s_rowl[8][16];                // per-wave row sums

  const int tid  = threadIdx.x;
  const int wv   = tid >> 5;        // wave 0..7
  const int lane = tid & 31;
  const int g    = wv >> 2;         // group 0/1
  const int hh   = wv & 3;          // head
  const int tg   = tid & 127;       // thread-in-group

  const int groupIdx = blockIdx.x * 2 + g;     // 0..255
  const int b  = groupIdx >> 6;
  const int k0 = (groupIdx & 63) * 16;

  const __bf16* WhT_bh = WhTbf + (size_t)(b * Hc + hh) * Dc * Mc;
  const float*  Wh1_bh = Wh1 + (size_t)(b * Hc + hh) * Mc;
  const float*  Wh2_bh = Wh2 + (size_t)(b * Hc + hh) * Mc;

  float we[EDc];
  #pragma unroll
  for (int e = 0; e < EDc; ++e) we[e] = w_e[hh * EDc + e];

  float wh1r[16];                   // loop-invariant row biases
  #pragma unroll
  for (int k = 0; k < 16; ++k) wh1r[k] = Wh1_bh[k0 + k];

  // --- per-thread streaming pointers: advance by constant stride per chunk --
  const char* srcp[8];
  #pragma unroll
  for (int j = 0; j < 8; ++j) {
    int f = tg + j * 128;           // float4 index; k = f>>6 (+2 per j),
    int k = f >> 6;                 // rem constant across j
    int rem = f & 63;
    int q = rem >> 1;
    int h4 = rem & 1;
    srcp[j] = (const char*)(comp_val +
                            (((size_t)b * Mc + (k0 + k)) * Mc + q) * EDc + h4 * 4);
  }
  const char* maskp = (const char*)(mch_mask +
                                    ((size_t)b * Mc + (k0 + (tg >> 3))) * Mc +
                                    (tg & 7) * 4);

  v8f acc[4] = {};                  // 16x64 f32 accumulators (4 N-tiles)
  float mrow = NEGV, lrow = 0.f;    // per-row softmax state (row = lane&15,
                                    // duplicated in both half-wave lanes)
  const int rowoff = (lane >> 4) * 8;
  const int nlane  = lane & 15;
  const int arow   = lane & 15;
  const int half   = lane >> 4;
  const int kb     = half * 8;

  // issue the 9 async copies for the next chunk into buffer `buf`
  auto issue_chunk = [&](int buf) {
    #pragma unroll
    for (int j = 0; j < 8; ++j) {
      int f = tg + j * 128;
      copy16_g2lds(srcp[j], &s_comp[g][buf][f * 4]);
      srcp[j] += 32 * EDc * 4;      // next q-chunk: +1024B
    }
    copy16_g2lds(maskp, &s_mask[g][buf][(tg >> 3) * 32 + (tg & 7) * 4]);
    maskp += 32 * 4;                // next q-chunk: +128B
  };

  issue_chunk(0);                   // prologue: chunk 0 in flight

  for (int i = 0; i < NCHUNK; ++i) {
    const int p = i & 1;
    const int q0 = i * 32;
    if (i + 1 < NCHUNK) {
      issue_chunk(1 - p);           // stream chunk i+1 while consuming chunk i
      wait_async_le<9>();           // in-order completion: chunk i is done
    } else {
      wait_async_le<0>();
    }
    __syncthreads();

    // ---- Phase B1: e = Wh1[k] + Wh2[q] + comp.we, leaky, mask ----
    {
      const float wh2q = Wh2_bh[q0 + lane];    // lane <-> q column
      float* pw = s_p[wv];
      const float* cbase = s_comp[g][p];
      const int* mbase = s_mask[g][p];
      #pragma unroll 4
      for (int k = 0; k < 16; ++k) {
        const float* cp = &cbase[(k * 32 + lane) * EDc];
        float dotv = 0.f;
        #pragma unroll
        for (int e = 0; e < EDc; ++e) dotv += cp[e] * we[e];
        float ev = wh1r[k] + wh2q + dotv;
        ev = ev > 0.f ? ev : ALPHA * ev;
        ev = (mbase[k * 32 + lane] > 0) ? ev : NEGV;
        pw[k * PSTR + lane] = ev;
      }
    }
    asm volatile("" ::: "memory");             // order LDS phases within wave

    // ---- Phase B2: online softmax; row = lane&15, each half-wave scans 16
    //      q values, partials combined across half-waves via shfl_xor ----
    {
      float* pr = &s_p[wv][arow * PSTR + half * 16];
      float rmax = pr[0];
      #pragma unroll
      for (int q = 1; q < 16; ++q) rmax = fmaxf(rmax, pr[q]);
      rmax = fmaxf(rmax, __shfl_xor(rmax, 16, 32));       // full-row max
      float mnew = fmaxf(mrow, rmax);
      float corr = __expf(mrow - mnew);
      float lsum = 0.f;
      #pragma unroll
      for (int q = 0; q < 16; ++q) {
        float pv = __expf(pr[q] - mnew);
        pr[q] = pv;
        lsum += pv;
      }
      lsum += __shfl_xor(lsum, 16, 32);                   // full-row sum
      lrow = lrow * corr + lsum;
      mrow = mnew;
      if (lane < 16) s_scale[wv][lane] = corr;
    }
    asm volatile("" ::: "memory");

    // ---- Phase B3: rescale accumulators (C row = j + rowoff) ----
    #pragma unroll
    for (int j = 0; j < 8; ++j) {
      float sc = s_scale[wv][rowoff + j];
      #pragma unroll
      for (int t = 0; t < 4; ++t) acc[t][j] *= sc;
    }

    // ---- Phase B4: pack P -> bf16 A fragment (16x32, contiguous k = 16*half + s) --
    v16bf Af;
    {
      const float* pw = s_p[wv];
      #pragma unroll
      for (int i2 = 0; i2 < 8; ++i2) {
        int K = half * 16 + 2 * i2;
        Af[2 * i2]     = (__bf16)pw[arow * PSTR + K];
        Af[2 * i2 + 1] = (__bf16)pw[arow * PSTR + K + 1];
      }
    }

    // ---- Phase B5: B fragments = raw bf16 b128 loads + 4x WMMA ----
    {
      const int KH = half * 16;
      #pragma unroll
      for (int t = 0; t < 4; ++t) {
        v16bf Bf = *(const v16bf*)(WhT_bh + (size_t)(t * 16 + nlane) * Mc + q0 + KH);
        acc[t] = __builtin_amdgcn_wmma_f32_16x16x32_bf16(
            false, Af, false, Bf, (short)0, acc[t], false, false);
      }
    }
    __syncthreads();   // all waves done with buf p before it is re-streamed
  }

  // ---- Finalize: divide by row sum, write out[b, m, h*64 + d] ----
  if (lane < 16) s_rowl[wv][lane] = lrow;
  asm volatile("" ::: "memory");
  float invl[8];
  #pragma unroll
  for (int j = 0; j < 8; ++j) invl[j] = 1.0f / s_rowl[wv][rowoff + j];
  #pragma unroll
  for (int j = 0; j < 8; ++j) {
    int m = k0 + rowoff + j;
    float* op = out + ((size_t)b * Mc + m) * (Hc * Dc) + hh * Dc;
    #pragma unroll
    for (int t = 0; t < 4; ++t) op[t * 16 + nlane] = acc[t][j] * invl[j];
  }
}

// ---------------------------------------------------------------------------
extern "C" void kernel_launch(void* const* d_in, const int* in_sizes, int n_in,
                              void* d_out, int out_size, void* d_ws, size_t ws_size,
                              hipStream_t stream) {
  (void)in_sizes; (void)n_in; (void)out_size; (void)ws_size;
  const float* hfeat  = (const float*)d_in[0];
  const int*   mch    = (const int*)d_in[1];
  const float* comp   = (const float*)d_in[2];
  const float* W      = (const float*)d_in[3];
  const float* W_edge = (const float*)d_in[4];
  const float* avec   = (const float*)d_in[5];
  float* out = (float*)d_out;

  // Workspace layout: bf16 WhT (2MB, 32B-aligned base), then f32 arrays.
  __bf16* WhTbf = (__bf16*)d_ws;                           // B*H*D*M bf16
  float*  wsf   = (float*)((char*)d_ws +
                           (size_t)Bc * Hc * Dc * Mc * sizeof(__bf16));
  float* Wh1 = wsf;                                        // 16K f32
  float* Wh2 = Wh1 + (size_t)Bc * Hc * Mc;                 // 16K f32
  float* w_e = Wh2 + (size_t)Bc * Hc * Mc;                 // 32 f32

  precompute_kernel<<<(Bc * Hc * Mc + 255) / 256, 256, 0, stream>>>(
      hfeat, W, W_edge, avec, WhTbf, Wh1, Wh2, w_e);

  attn_main_kernel<<<(Bc * (Mc / 16)) / 2, 256, 0, stream>>>(
      comp, mch, WhTbf, Wh1, Wh2, w_e, out);
}
